// TriangleAttention_53944789238378
// MI455X (gfx1250) — matrix operand order repaired and checked by reading, not compiled, not run on hardware
//
#include <hip/hip_runtime.h>
#include <hip/hip_bf16.h>

// ---------------------------------------------------------------------------
// TriangleAttention for MI455X (gfx1250, wave32, WMMA).
//   phase 0: pack  z -> bf16 row-major; transpose Wq/Wk/Wv/Wg/Wo -> bf16 N-major
//   phase 1: fused QKVG GEMM  [8192x256]x[256x2048] via v_wmma_f32_16x16x32_bf16
//            epilogue: q *= 1/sqrt(64);  gate = sigmoid(sigmoid(x + bg))
//   phase 2: gather attention (wave per (edge, head); coalesced 64-wide rows)
//   phase 3: output GEMM [8192x512]x[512x256] + bo via WMMA, fp32 out
// ---------------------------------------------------------------------------

#define D_PAIR   256
#define HEAD_DIM 64
#define NUM_HEADS 8
#define D_HID    512
#define N_EDGE   8192
#define N_KV     16

typedef __attribute__((ext_vector_type(16))) __bf16 v16bf;
typedef __attribute__((ext_vector_type(8)))  __bf16 bf16x8;
typedef __attribute__((ext_vector_type(8)))  float  v8f;

__device__ __forceinline__ unsigned short f32_bf16(float f) {
    unsigned u; __builtin_memcpy(&u, &f, 4);
    unsigned r = u + 0x7FFFu + ((u >> 16) & 1u);   // round-to-nearest-even
    return (unsigned short)(r >> 16);
}
__device__ __forceinline__ float lo_bf(unsigned u) {
    unsigned b = u << 16; float f; __builtin_memcpy(&f, &b, 4); return f;
}
__device__ __forceinline__ float hi_bf(unsigned u) {
    unsigned b = u & 0xFFFF0000u; float f; __builtin_memcpy(&f, &b, 4); return f;
}
__device__ __forceinline__ float sigmoidf(float x) { return 1.f / (1.f + __expf(-x)); }

union FragU { v16bf v; bf16x8 h[2]; };

// One wave computes one 16x16 f32 tile of A[MxK] * Bt[NxK]^T (both bf16).
// A row-major [*, KDIM]; Bt N-major [*, KDIM].  KDIM % 32 == 0.
template <int KDIM>
__device__ __forceinline__ v8f wmma_tile(const unsigned short* __restrict__ A,
                                         const unsigned short* __restrict__ Bt,
                                         int m0, int n0) {
    const int lane = threadIdx.x & 31;
    const int sub  = lane & 15;
    const int half = lane >> 4;
    // A frag: lane=M, K pattern = {h*8 .. h*8+7} U {h*8+16 .. h*8+23}
    const unsigned short* pa = A  + (size_t)(m0 + sub) * KDIM + half * 8;
    // B frag: lane=N, K pattern = {h*8 .. h*8+7} U {h*8+16 .. h*8+23}
    const unsigned short* pb = Bt + (size_t)(n0 + sub) * KDIM + half * 8;
    v8f acc = {0.f, 0.f, 0.f, 0.f, 0.f, 0.f, 0.f, 0.f};
#pragma unroll
    for (int kt = 0; kt < KDIM; kt += 32) {
        FragU a, b;
        a.h[0] = *(const bf16x8*)(pa + kt);
        a.h[1] = *(const bf16x8*)(pa + kt + 16);
        b.h[0] = *(const bf16x8*)(pb + kt);
        b.h[1] = *(const bf16x8*)(pb + kt + 16);
        acc = __builtin_amdgcn_wmma_f32_16x16x32_bf16(
                false, a.v, false, b.v, (short)0, acc, false, false);
    }
    return acc;
}

// ---------------- phase 0: packing ----------------
__global__ __launch_bounds__(256) void cvt_bf16_kernel(const float* __restrict__ src,
                                                       unsigned short* __restrict__ dst, int n) {
    int i = blockIdx.x * 256 + threadIdx.x;
    if (i < n) dst[i] = f32_bf16(src[i]);
}

// src[K][N] row-major fp32  ->  dst[N][K] bf16
__global__ __launch_bounds__(256) void transpose_bf16_kernel(const float* __restrict__ src,
                                                             unsigned short* __restrict__ dst,
                                                             int K, int N) {
    int i = blockIdx.x * 256 + threadIdx.x;
    if (i < K * N) {
        int n = i / K, k = i - n * K;
        dst[i] = f32_bf16(src[(size_t)k * N + n]);
    }
}

// ---------------- phase 1: fused QKVG GEMM ----------------
// grid (16, 512), block 256 (8 waves). N_total = 4*512 = 2048.
__global__ __launch_bounds__(256) void qkvg_gemm_kernel(const unsigned short* __restrict__ zb,
                                                        const unsigned short* __restrict__ WT,
                                                        const float* __restrict__ bg,
                                                        unsigned short* __restrict__ qb,
                                                        unsigned short* __restrict__ kb,
                                                        unsigned short* __restrict__ vb,
                                                        unsigned short* __restrict__ gate) {
    const int m0 = blockIdx.y * 16;
    const int n0 = blockIdx.x * 128 + (threadIdx.x >> 5) * 16;
    v8f acc = wmma_tile<D_PAIR>(zb, WT, m0, n0);

    const int lane   = threadIdx.x & 31;
    const int col    = lane & 15;
    const int rowoff = (lane >> 4) * 8;
    const int nglob  = n0 + col;
    const int which  = nglob >> 9;         // 0=q 1=k 2=v 3=g (wave-uniform)
    const int ncol   = nglob & 511;
    unsigned short* dst = (which == 0) ? qb : (which == 1) ? kb : (which == 2) ? vb : gate;
    const float bgv = (which == 3) ? bg[ncol] : 0.f;
#pragma unroll
    for (int r = 0; r < 8; r++) {
        int m = m0 + rowoff + r;
        float x = acc[r];
        if (which == 0) x *= 0.125f;                       // HEAD_DIM^-0.5
        if (which == 3) x = sigmoidf(sigmoidf(x + bgv));   // double sigmoid (faithful)
        dst[(size_t)m * D_HID + ncol] = f32_bf16(x);
    }
}

// ---------------- phase 2: gather attention ----------------
// grid 8192 (one block per edge), block 256: wave = head, lane covers 2 dims.
__global__ __launch_bounds__(256) void attn_gather_kernel(const int* __restrict__ eil,
                                                          const unsigned short* __restrict__ qb,
                                                          const unsigned short* __restrict__ kb,
                                                          const unsigned short* __restrict__ vb,
                                                          const unsigned short* __restrict__ gate,
                                                          unsigned short* __restrict__ obuf) {
    const int e    = blockIdx.x;
    const int h    = threadIdx.x >> 5;
    const int lane = threadIdx.x & 31;
    const int off  = h * HEAD_DIM + lane * 2;
    const int* idx = eil + (size_t)e * N_KV * 2;

    float o0 = 0.f, o1 = 0.f;
#pragma unroll
    for (int j = 0; j < N_KV; j++) {
        const int t = idx[2 * j + 0];   // target -> q
        const int s = idx[2 * j + 1];   // source -> k,v
        unsigned qu = *(const unsigned*)(qb + (size_t)t * D_HID + off);
        unsigned ku = *(const unsigned*)(kb + (size_t)s * D_HID + off);
        float p = lo_bf(qu) * lo_bf(ku) + hi_bf(qu) * hi_bf(ku);
#pragma unroll
        for (int m = 16; m >= 1; m >>= 1) p += __shfl_xor(p, m, 32);
        unsigned vu = *(const unsigned*)(vb + (size_t)s * D_HID + off);
        o0 += p * lo_bf(vu);
        o1 += p * hi_bf(vu);
    }
    unsigned gu = *(const unsigned*)(gate + (size_t)e * D_HID + off);
    o0 *= lo_bf(gu);
    o1 *= hi_bf(gu);
    unsigned packed = (unsigned)f32_bf16(o0) | ((unsigned)f32_bf16(o1) << 16);
    *(unsigned*)(obuf + (size_t)e * D_HID + off) = packed;
}

// ---------------- phase 3: output GEMM + bias ----------------
// grid (2, 512), block 256.
__global__ __launch_bounds__(256) void out_gemm_kernel(const unsigned short* __restrict__ ob,
                                                       const unsigned short* __restrict__ WoT,
                                                       const float* __restrict__ bo,
                                                       float* __restrict__ out) {
    const int m0 = blockIdx.y * 16;
    const int n0 = blockIdx.x * 128 + (threadIdx.x >> 5) * 16;
    v8f acc = wmma_tile<D_HID>(ob, WoT, m0, n0);

    const int lane   = threadIdx.x & 31;
    const int col    = lane & 15;
    const int rowoff = (lane >> 4) * 8;
    const float bias = bo[n0 + col];
#pragma unroll
    for (int r = 0; r < 8; r++) {
        int m = m0 + rowoff + r;
        out[(size_t)m * D_PAIR + n0 + col] = acc[r] + bias;
    }
}

// ---------------------------------------------------------------------------
extern "C" void kernel_launch(void* const* d_in, const int* in_sizes, int n_in,
                              void* d_out, int out_size, void* d_ws, size_t ws_size,
                              hipStream_t stream) {
    const float* z  = (const float*)d_in[0];
    const int*   eil = (const int*)d_in[1];
    // d_in[2] = edge_index_right (unused by reference)
    const float* Wq = (const float*)d_in[3];
    const float* Wk = (const float*)d_in[4];
    const float* Wv = (const float*)d_in[5];
    const float* Wg = (const float*)d_in[6];
    const float* bg = (const float*)d_in[7];
    const float* Wo = (const float*)d_in[8];
    const float* bo = (const float*)d_in[9];

    // workspace carve-up (bf16 elements; all chunks 16B aligned); ~47.4 MB total
    unsigned short* ws16 = (unsigned short*)d_ws;
    size_t o = 0;
    unsigned short* zb   = ws16 + o; o += (size_t)N_EDGE * D_PAIR;      // 4 MB
    unsigned short* WT   = ws16 + o; o += (size_t)4 * D_HID * D_PAIR;   // 1 MB (q|k|v|g transposed)
    unsigned short* WoT  = ws16 + o; o += (size_t)D_PAIR * D_HID;       // 0.25 MB
    unsigned short* qb   = ws16 + o; o += (size_t)N_EDGE * D_HID;       // 8 MB
    unsigned short* kb   = ws16 + o; o += (size_t)N_EDGE * D_HID;
    unsigned short* vb   = ws16 + o; o += (size_t)N_EDGE * D_HID;
    unsigned short* gate = ws16 + o; o += (size_t)N_EDGE * D_HID;
    unsigned short* ob   = ws16 + o; o += (size_t)N_EDGE * D_HID;

    // phase 0: pack
    {
        int n = N_EDGE * D_PAIR;
        cvt_bf16_kernel<<<(n + 255) / 256, 256, 0, stream>>>(z, zb, n);
        int nw = D_PAIR * D_HID;   // 256*512
        int g  = (nw + 255) / 256;
        transpose_bf16_kernel<<<g, 256, 0, stream>>>(Wq, WT + 0 * (size_t)D_HID * D_PAIR, D_PAIR, D_HID);
        transpose_bf16_kernel<<<g, 256, 0, stream>>>(Wk, WT + 1 * (size_t)D_HID * D_PAIR, D_PAIR, D_HID);
        transpose_bf16_kernel<<<g, 256, 0, stream>>>(Wv, WT + 2 * (size_t)D_HID * D_PAIR, D_PAIR, D_HID);
        transpose_bf16_kernel<<<g, 256, 0, stream>>>(Wg, WT + 3 * (size_t)D_HID * D_PAIR, D_PAIR, D_HID);
        transpose_bf16_kernel<<<g, 256, 0, stream>>>(Wo, WoT, D_HID, D_PAIR);
    }
    // phase 1: fused QKVG GEMM  (N_total = 2048 = 16 blocks * 8 waves * 16)
    qkvg_gemm_kernel<<<dim3(16, N_EDGE / 16), 256, 0, stream>>>(zb, WT, bg, qb, kb, vb, gate);
    // phase 2: gather attention
    attn_gather_kernel<<<N_EDGE, 256, 0, stream>>>(eil, qb, kb, vb, gate, ob);
    // phase 3: output GEMM (N = 256 = 2 blocks * 8 waves * 16)
    out_gemm_kernel<<<dim3(2, N_EDGE / 16), 256, 0, stream>>>(ob, WoT, bo, (float*)d_out);
}
